// PatternCWanRoPE_81776177315984
// MI455X (gfx1250) — matrix operand order repaired and checked by reading, not compiled, not run on hardware
//
#include <hip/hip_runtime.h>

typedef __attribute__((ext_vector_type(16))) _Float16 v16h;
typedef __attribute__((ext_vector_type(8)))  _Float16 v8h;
typedef __attribute__((ext_vector_type(2)))  __fp16   v2fp16;
typedef __attribute__((ext_vector_type(8)))  float    v8f;
typedef __attribute__((ext_vector_type(4)))  float    v4f;

#define B_ 2
#define S_ 2048
#define H_ 16
#define D_ 128
#define SCALE_ 0.08838834764831845f   // 1/sqrt(128)
#define NEGINF_ (-1.0e30f)

// ---------------------------------------------------------------------------
// Kernel 1: interleaved RoPE on q,k (1/sqrt(D) folded into q), convert to f16,
// pack Qh/Kh as [B*H][S][D] row-major and V transposed as Vt [B*H][D][S] so all
// WMMA fragment loads in the attention kernel are contiguous 16B b128 loads.
// ---------------------------------------------------------------------------
__global__ __launch_bounds__(256) void rope_pack_kernel(
    const float* __restrict__ q, const float* __restrict__ k,
    const float* __restrict__ v, const float* __restrict__ freqs,
    _Float16* __restrict__ Qh, _Float16* __restrict__ Kh,
    _Float16* __restrict__ Vt)
{
    const int DP = D_ / 2;
    int idx = blockIdx.x * blockDim.x + threadIdx.x;     // over B*S*H*(D/2)
    if (idx >= B_ * S_ * H_ * DP) return;
    int i = idx % DP;
    int h = (idx / DP) % H_;
    int s = (idx / (DP * H_)) % S_;
    int b =  idx / (DP * H_ * S_);

    const float2 qp = ((const float2*)q)[idx];
    const float2 kp = ((const float2*)k)[idx];
    const float2 vp = ((const float2*)v)[idx];
    const float2 fp = ((const float2*)freqs)[s * DP + i];

    float rq0 = (fp.x * qp.x - fp.y * qp.y) * SCALE_;
    float rq1 = (fp.y * qp.x + fp.x * qp.y) * SCALE_;
    float rk0 =  fp.x * kp.x - fp.y * kp.y;
    float rk1 =  fp.y * kp.x + fp.x * kp.y;

    size_t row = ((size_t)(b * H_ + h) * S_ + s) * D_ + 2 * i;
    Qh[row]     = (_Float16)rq0;
    Qh[row + 1] = (_Float16)rq1;
    Kh[row]     = (_Float16)rk0;
    Kh[row + 1] = (_Float16)rk1;

    size_t vt = ((size_t)(b * H_ + h) * D_ + 2 * i) * S_ + s;
    Vt[vt]      = (_Float16)vp.x;
    Vt[vt + S_] = (_Float16)vp.y;
}

// A-fragment (16x32 f16): lane m = lane%16; halves j -> K = hi*16+j (contiguous),
// Two contiguous 16B loads.
__device__ __forceinline__ v16h load_a_frag(const _Float16* __restrict__ base,
                                            int off_lo, int off_hi)
{
    v8h a = *(const v8h*)(base + off_lo);
    v8h b = *(const v8h*)(base + off_hi);
    return __builtin_shufflevector(a, b, 0,1,2,3,4,5,6,7,8,9,10,11,12,13,14,15);
}

// v_cvt_pk_f16_f32: pack two floats into one dword of f16 halves (x in [15:0])
__device__ __forceinline__ unsigned int pk2u(float x, float y)
{
    union { v2fp16 h; unsigned int u; } c;
    c.h = __builtin_amdgcn_cvt_pkrtz(x, y);
    return c.u;
}

// ---------------------------------------------------------------------------
// Kernel 2: one wave32 per 32-row q tile (two 16-column q groups sharing every
// K/V fragment). Transposed flash attention:
//   S^T = K_tile * Q^T   (v_wmma_f32_16x16x32_f16, matmul-K = head dim)
//   O^T += V^T * P^T     (P^T B-fragment built from score C-regs, 4 packed
//                         dword shuffles per group)
// Per-lane online softmax state (m,l): each lane owns one q column per group.
// ---------------------------------------------------------------------------
__global__ __launch_bounds__(256) void fa_fwd_kernel(
    const _Float16* __restrict__ Qh, const _Float16* __restrict__ Kh,
    const _Float16* __restrict__ Vt, float* __restrict__ out)
{
    const int TPS   = S_ / 32;                       // 64 q tiles per (b,h)
    const int wave  = blockIdx.x * (blockDim.x >> 5) + (threadIdx.x >> 5);
    const int lane  = threadIdx.x & 31;
    const int bh    = wave / TPS;
    const int qbase = (wave % TPS) * 32;
    const int lid   = lane & 15;
    const int hi    = lane >> 4;

    const _Float16* Kbase = Kh + (size_t)bh * S_ * D_;
    const _Float16* Vbase = Vt + (size_t)bh * D_ * S_;

    int qrow[2] = { qbase + lid, qbase + 16 + lid };

    // Q as B operand: lane q-col = lid; halves j -> d = dc*32 + hi*16 + j
    v16h qb[2][4];
#pragma unroll
    for (int g = 0; g < 2; ++g) {
        const _Float16* Qrow = Qh + ((size_t)bh * S_ + qrow[g]) * D_;
#pragma unroll
        for (int dc = 0; dc < 4; ++dc) {
            v8h a = *(const v8h*)(Qrow + dc * 32 + hi * 16);
            v8h b = *(const v8h*)(Qrow + dc * 32 + hi * 16 + 8);
            qb[g][dc] = __builtin_shufflevector(a, b,
                0,1,2,3,4,5,6,7,8,9,10,11,12,13,14,15);
        }
    }

    v8f ot[2][8];                                    // O^T accumulators
#pragma unroll
    for (int g = 0; g < 2; ++g)
#pragma unroll
        for (int dm = 0; dm < 8; ++dm) ot[g][dm] = (v8f)0.0f;
    float m_run[2] = { NEGINF_, NEGINF_ };
    float l_run[2] = { 0.0f, 0.0f };

    const int nkt = (qbase >> 5) + 1;                // causal: tiles up to qbase
    for (int kt = 0; kt < nkt; ++kt) {
        const int ktbase = kt << 5;

        // ---- S^T tiles; each K fragment feeds both q groups ----
        v8f st[2][2];
        st[0][0] = (v8f)0.0f; st[0][1] = (v8f)0.0f;
        st[1][0] = (v8f)0.0f; st[1][1] = (v8f)0.0f;
#pragma unroll
        for (int mc = 0; mc < 2; ++mc) {
            const _Float16* Krow = Kbase + ((size_t)(ktbase + mc * 16 + lid)) * D_;
#pragma unroll
            for (int dc = 0; dc < 4; ++dc) {
                v16h ka = load_a_frag(Krow, dc * 32 + hi * 16, dc * 32 + hi * 16 + 8);
                st[0][mc] = __builtin_amdgcn_wmma_f32_16x16x32_f16(
                    false, ka, false, qb[0][dc], (short)0, st[0][mc], false, false);
                st[1][mc] = __builtin_amdgcn_wmma_f32_16x16x32_f16(
                    false, ka, false, qb[1][dc], (short)0, st[1][mc], false, false);
            }
        }

        // ---- causal mask: only the last tile (ktbase == qbase) is partial ----
        if (kt == nkt - 1) {
#pragma unroll
            for (int g = 0; g < 2; ++g)
#pragma unroll
                for (int mc = 0; mc < 2; ++mc)
#pragma unroll
                    for (int vv = 0; vv < 8; ++vv) {
                        int kk = ktbase + mc * 16 + vv + 8 * hi;
                        if (kk > qrow[g]) st[g][mc][vv] = NEGINF_;
                    }
        }

        // ---- per-lane online softmax + packed P^T fragment per group ----
        v16h pf[2];
#pragma unroll
        for (int g = 0; g < 2; ++g) {
            float tmax = NEGINF_;
#pragma unroll
            for (int vv = 0; vv < 8; ++vv)
                tmax = fmaxf(tmax, fmaxf(st[g][0][vv], st[g][1][vv]));
            tmax = fmaxf(tmax, __shfl_xor(tmax, 16, 32));
            float mnew  = fmaxf(m_run[g], tmax);
            float alpha = __expf(m_run[g] - mnew);

            float p0[8], p1[8], psum = 0.0f;
#pragma unroll
            for (int vv = 0; vv < 8; ++vv) {
                p0[vv] = __expf(st[g][0][vv] - mnew);
                p1[vv] = __expf(st[g][1][vv] - mnew);
                psum += p0[vv] + p1[vv];
            }
            psum += __shfl_xor(psum, 16, 32);
            l_run[g] = l_run[g] * alpha + psum;
            m_run[g] = mnew;
#pragma unroll
            for (int dm = 0; dm < 8; ++dm)
#pragma unroll
                for (int vv = 0; vv < 8; ++vv) ot[g][dm][vv] *= alpha;

            // pack to f16 pairs; exchange only what the partner needs:
            //   lo8: hi=0 -> own p0 ; hi=1 -> partner p1
            //   hi8: hi=0 -> partner p0 ; hi=1 -> own p1
            union { unsigned int u[8]; v16h v; } pfu;
#pragma unroll
            for (int i = 0; i < 4; ++i) {
                unsigned int a = pk2u(p0[2 * i], p0[2 * i + 1]);   // own p0 pack
                unsigned int b = pk2u(p1[2 * i], p1[2 * i + 1]);   // own p1 pack
                unsigned int x = hi ? a : b;                       // value partner needs
                unsigned int y = __shfl_xor(x, 16, 32);            // partner's value
                pfu.u[i]     = hi ? y : a;
                pfu.u[i + 4] = hi ? b : y;
            }
            pf[g] = pfu.v;
        }

        // ---- O^T += V^T * P^T ; each V fragment feeds both q groups ----
#pragma unroll
        for (int dm = 0; dm < 8; ++dm) {
            const _Float16* Vrow = Vbase + ((size_t)(dm * 16 + lid)) * S_;
            v16h va = load_a_frag(Vrow, ktbase + hi * 16, ktbase + hi * 16 + 8);
            ot[0][dm] = __builtin_amdgcn_wmma_f32_16x16x32_f16(
                false, va, false, pf[0], (short)0, ot[0][dm], false, false);
            ot[1][dm] = __builtin_amdgcn_wmma_f32_16x16x32_f16(
                false, va, false, pf[1], (short)0, ot[1][dm], false, false);
        }
    }

    // ---- epilogue: normalize, store to out[b][h][q][d] (d contiguous) ----
#pragma unroll
    for (int g = 0; g < 2; ++g) {
        float invl = 1.0f / l_run[g];
        float* orow = out + ((size_t)bh * S_ + qrow[g]) * D_;
#pragma unroll
        for (int dm = 0; dm < 8; ++dm) {
            int d0 = dm * 16 + 8 * hi;
            v4f lo4 = { ot[g][dm][0] * invl, ot[g][dm][1] * invl,
                        ot[g][dm][2] * invl, ot[g][dm][3] * invl };
            v4f hi4 = { ot[g][dm][4] * invl, ot[g][dm][5] * invl,
                        ot[g][dm][6] * invl, ot[g][dm][7] * invl };
            *(v4f*)(orow + d0)     = lo4;
            *(v4f*)(orow + d0 + 4) = hi4;
        }
    }
}

extern "C" void kernel_launch(void* const* d_in, const int* in_sizes, int n_in,
                              void* d_out, int out_size, void* d_ws, size_t ws_size,
                              hipStream_t stream)
{
    (void)in_sizes; (void)n_in; (void)out_size; (void)ws_size;
    const float* q = (const float*)d_in[0];
    const float* k = (const float*)d_in[1];
    const float* v = (const float*)d_in[2];
    const float* f = (const float*)d_in[3];
    float* out = (float*)d_out;

    const size_t n = (size_t)B_ * H_ * S_ * D_;
    _Float16* Qh = (_Float16*)d_ws;      // 16 MB
    _Float16* Kh = Qh + n;               // 16 MB
    _Float16* Vt = Kh + n;               // 16 MB (transposed [bh][d][s])

    int total = B_ * S_ * H_ * (D_ / 2);
    rope_pack_kernel<<<(total + 255) / 256, 256, 0, stream>>>(q, k, v, f, Qh, Kh, Vt);

    int waves = B_ * H_ * (S_ / 32);     // 2048 q tiles of 32 rows
    fa_fwd_kernel<<<waves / 8, 256, 0, stream>>>(Qh, Kh, Vt, out);
}
